// AttentionBlock_13718125543864
// MI455X (gfx1250) — matrix operand order repaired and checked by reading, not compiled, not run on hardware
//
#include <hip/hip_runtime.h>

// ---------------------------------------------------------------------------
// Attention block for MI455X (gfx1250, wave32, WMMA).
// All contractions use v_wmma_f32_16x16x32_bf16 (bf16 operands, f32 accum).
// Softmax / RMSNorm / RoPE stay in fp32.
// ---------------------------------------------------------------------------

#define Bc 2
#define Tc 2048
#define Cc 1024
#define Hc 16
#define Dc 64

typedef __attribute__((ext_vector_type(16))) __bf16 bf16x16;
typedef __attribute__((ext_vector_type(8)))  __bf16 bf16x8;
typedef __attribute__((ext_vector_type(8)))  float  f32x8;

// ---- WMMA fragment helpers (contiguous k order k = 16*(lane>=16) + e, same for A and B) ------

// A fragment, 16x32 bf16, source row-major [16][K..K+31], stride ld elements.
// lane: M = lane&15; element e -> K = khalf+e,
// khalf = (lane>=16)*16.  Two contiguous 16B loads per lane.
__device__ __forceinline__ bf16x16 load_a_frag(const __bf16* base, int ld, int lane) {
  const __bf16* p = base + (size_t)(lane & 15) * ld + ((lane >> 4) << 4);
  bf16x8 lo = *(const bf16x8*)(p);
  bf16x8 hi = *(const bf16x8*)(p + 8);
  return __builtin_shufflevector(lo, hi, 0,1,2,3,4,5,6,7,8,9,10,11,12,13,14,15);
}

// B fragment, 32x16 bf16 where b[e] = W[n = lane&15][k = e + (lane>=16)*16].
// For all contractions here W rows are K-contiguous -> one 32B load per lane.
__device__ __forceinline__ bf16x16 load_b_frag(const __bf16* base, int ld, int lane) {
  const __bf16* p = base + (size_t)(lane & 15) * ld + ((lane >> 4) << 4);
  return *(const bf16x16*)p;
}

__device__ __forceinline__ f32x8 wmma_bf16(bf16x16 a, bf16x16 b, f32x8 c) {
  return __builtin_amdgcn_wmma_f32_16x16x32_bf16(false, a, false, b, (short)0, c,
                                                 false, false);
}

// ---- Kernel 1: RMSNorm + cast to bf16 --------------------------------------
__global__ __launch_bounds__(256) void rmsnorm_bf16_kernel(
    const float* __restrict__ x, const float* __restrict__ w,
    __bf16* __restrict__ xn) {
  __shared__ float sred[8];
  const int row = blockIdx.x;
  const int t = threadIdx.x;
  const float4 v = ((const float4*)(x + (size_t)row * Cc))[t];
  float ss = v.x * v.x + v.y * v.y + v.z * v.z + v.w * v.w;
  #pragma unroll
  for (int off = 16; off; off >>= 1) ss += __shfl_xor(ss, off, 32);
  if ((t & 31) == 0) sred[t >> 5] = ss;
  __syncthreads();
  float tot = 0.f;
  #pragma unroll
  for (int i = 0; i < 8; ++i) tot += sred[i];
  const float sc = rsqrtf(tot * (1.0f / (float)Cc) + 1e-5f);
  const float4 wv = ((const float4*)w)[t];
  __bf16* o = xn + (size_t)row * Cc + t * 4;
  o[0] = (__bf16)(v.x * sc * wv.x);
  o[1] = (__bf16)(v.y * sc * wv.y);
  o[2] = (__bf16)(v.z * sc * wv.z);
  o[3] = (__bf16)(v.w * sc * wv.w);
}

// ---- Kernel 2: fp32 -> bf16 weight cast ------------------------------------
__global__ __launch_bounds__(256) void cvt_f32_bf16_kernel(
    const float* __restrict__ in, __bf16* __restrict__ out, int n) {
  const int i = (blockIdx.x * 256 + threadIdx.x) * 4;
  if (i < n) {
    const float4 v = *(const float4*)(in + i);
    out[i + 0] = (__bf16)v.x;
    out[i + 1] = (__bf16)v.y;
    out[i + 2] = (__bf16)v.z;
    out[i + 3] = (__bf16)v.w;
  }
}

// ---- Kernel 3/6: GEMM  out[m][n] = sum_k A[m][k]*W[n][k] + bias[n] ---------
// Block = 8 waves; wave computes a 32x32 tile (2x2 WMMA frags); block = 128x64.
__global__ __launch_bounds__(256) void gemm_bf16_nt_kernel(
    const __bf16* __restrict__ A, const __bf16* __restrict__ W,
    const float* __restrict__ bias, float* __restrict__ out,
    int M, int N, int K) {
  const int lane = threadIdx.x & 31;
  const int wv = threadIdx.x >> 5;
  const int mbase = blockIdx.x * 128 + (wv >> 1) * 32;
  const int nbase = blockIdx.y * 64 + (wv & 1) * 32;
  const __bf16* a0p = A + (size_t)mbase * K;
  const __bf16* a1p = A + (size_t)(mbase + 16) * K;
  const __bf16* b0p = W + (size_t)nbase * K;
  const __bf16* b1p = W + (size_t)(nbase + 16) * K;
  f32x8 c00 = {}, c01 = {}, c10 = {}, c11 = {};
  for (int k0 = 0; k0 < K; k0 += 32) {
    __builtin_prefetch(a0p + k0 + 256);   // global_prefetch_b8
    __builtin_prefetch(b0p + k0 + 256);
    bf16x16 a0 = load_a_frag(a0p + k0, K, lane);
    bf16x16 a1 = load_a_frag(a1p + k0, K, lane);
    bf16x16 b0 = load_b_frag(b0p + k0, K, lane);
    bf16x16 b1 = load_b_frag(b1p + k0, K, lane);
    c00 = wmma_bf16(a0, b0, c00);
    c01 = wmma_bf16(a0, b1, c01);
    c10 = wmma_bf16(a1, b0, c10);
    c11 = wmma_bf16(a1, b1, c11);
  }
  const int g = lane >> 4, col = lane & 15;
  #pragma unroll
  for (int e = 0; e < 8; ++e) {
    const int r0 = mbase + e + 8 * g;
    const int r1 = r0 + 16;
    const int c0 = nbase + col;
    const int c1 = c0 + 16;
    out[(size_t)r0 * N + c0] = c00[e] + bias[c0];
    out[(size_t)r0 * N + c1] = c01[e] + bias[c1];
    out[(size_t)r1 * N + c0] = c10[e] + bias[c0];
    out[(size_t)r1 * N + c1] = c11[e] + bias[c1];
  }
}

// ---- Kernel 4: RoPE on Q,K + split/cast; V stored transposed ---------------
// qh,kh: [B][H][T][D] bf16 (d-contiguous).  vt: [B][H][D][T] bf16 (t-contig).
__global__ __launch_bounds__(256) void rope_split_kernel(
    const float* __restrict__ qkv, __bf16* __restrict__ qh,
    __bf16* __restrict__ kh, __bf16* __restrict__ vt) {
  const int bt = blockIdx.x;
  const int b = bt >> 11;          // T = 2048
  const int t = bt & (Tc - 1);
  const float* row = qkv + (size_t)bt * (3 * Cc);
  for (int j = threadIdx.x; j < Hc * 32; j += 256) {
    const int h = j >> 5, p = j & 31;
    const float invf = __powf(10000.0f, -(float)p * (1.0f / 32.0f));
    float sv, cv;
    __sincosf((float)t * invf, &sv, &cv);
    const size_t ob = ((size_t)(b * Hc + h) * Tc + t) * Dc;
    const float q0 = row[h * Dc + p], q1 = row[h * Dc + p + 32];
    qh[ob + p]      = (__bf16)(q0 * cv - q1 * sv);
    qh[ob + p + 32] = (__bf16)(q1 * cv + q0 * sv);
    const float k0 = row[Cc + h * Dc + p], k1 = row[Cc + h * Dc + p + 32];
    kh[ob + p]      = (__bf16)(k0 * cv - k1 * sv);
    kh[ob + p + 32] = (__bf16)(k1 * cv + k0 * sv);
  }
  for (int c = threadIdx.x; c < Cc; c += 256) {
    const int h = c >> 6, d = c & 63;
    vt[((size_t)(b * Hc + h) * Dc + d) * Tc + t] = (__bf16)row[2 * Cc + c];
  }
}

// ---- Kernel 5: causal flash attention --------------------------------------
// One wave per 16 query rows; key tiles of 32; online softmax; all matmuls
// via WMMA.  P restaged through per-wave LDS (intra-wave s_wait_dscnt only —
// causal loop trip counts diverge between waves, so no __syncthreads).
__global__ __launch_bounds__(256) void flash_attn_kernel(
    const __bf16* __restrict__ qh, const __bf16* __restrict__ kh,
    const __bf16* __restrict__ vt, __bf16* __restrict__ attn) {
  __shared__ __align__(32) __bf16 pst[8][16][32];
  const int lane = threadIdx.x & 31;
  const int wv = threadIdx.x >> 5;
  const int tilesPerHead = Tc / 128;                  // 16
  const int bh = blockIdx.x / tilesPerHead;
  const int tile = blockIdx.x % tilesPerHead;
  const int b = bh / Hc, h = bh % Hc;
  const int q0 = tile * 128 + wv * 16;
  const int g = lane >> 4, col = lane & 15;

  const __bf16* qbase = qh + (size_t)bh * Tc * Dc;
  const __bf16* kbase = kh + (size_t)bh * Tc * Dc;
  const __bf16* vbase = vt + (size_t)bh * Dc * Tc;

  const bf16x16 aq0 = load_a_frag(qbase + (size_t)q0 * Dc, Dc, lane);
  const bf16x16 aq1 = load_a_frag(qbase + (size_t)q0 * Dc + 32, Dc, lane);

  f32x8 o0 = {}, o1 = {}, o2 = {}, o3 = {};
  float m[8], l[8];
  #pragma unroll
  for (int e = 0; e < 8; ++e) { m[e] = -3.0e38f; l[e] = 0.0f; }

  const int qmax = q0 + 15;
  for (int kb = 0; kb <= qmax; kb += 32) {
    // Scores: S = (Q * K^T) / sqrt(D); K=64 -> 2 WMMAs per 16x16 tile.
    f32x8 s0 = {}, s1 = {};
    s0 = wmma_bf16(aq0, load_b_frag(kbase + (size_t)kb * Dc, Dc, lane), s0);
    s0 = wmma_bf16(aq1, load_b_frag(kbase + (size_t)kb * Dc + 32, Dc, lane), s0);
    s1 = wmma_bf16(aq0, load_b_frag(kbase + (size_t)(kb + 16) * Dc, Dc, lane), s1);
    s1 = wmma_bf16(aq1, load_b_frag(kbase + (size_t)(kb + 16) * Dc + 32, Dc, lane), s1);

    // Online softmax over the 32 keys of this tile.
    #pragma unroll
    for (int e = 0; e < 8; ++e) {
      const int row = q0 + e + 8 * g;
      float v0 = s0[e] * 0.125f;
      float v1 = s1[e] * 0.125f;
      if (kb + col > row)      v0 = -1.0e30f;
      if (kb + 16 + col > row) v1 = -1.0e30f;
      float mt = fmaxf(v0, v1);
      #pragma unroll
      for (int off = 1; off < 16; off <<= 1) mt = fmaxf(mt, __shfl_xor(mt, off, 16));
      const float mn = fmaxf(m[e], mt);
      const float alpha = __expf(m[e] - mn);
      const float p0 = __expf(v0 - mn);
      const float p1 = __expf(v1 - mn);
      float rs = p0 + p1;
      #pragma unroll
      for (int off = 1; off < 16; off <<= 1) rs += __shfl_xor(rs, off, 16);
      l[e] = l[e] * alpha + rs;
      m[e] = mn;
      o0[e] *= alpha; o1[e] *= alpha; o2[e] *= alpha; o3[e] *= alpha;
      pst[wv][e + 8 * g][col]      = (__bf16)p0;   // C-layout -> [row][col]
      pst[wv][e + 8 * g][col + 16] = (__bf16)p1;
    }

    // Intra-wave LDS fence (DS ops in-order per wave; no barrier needed).
    asm volatile("s_wait_dscnt 0" ::: "memory");

    // P (16x32) as A-fragment, V^T rows as B-fragments: O += P * V.
    const bf16x16 ap = load_a_frag(&pst[wv][0][0], 32, lane);
    o0 = wmma_bf16(ap, load_b_frag(vbase + kb, Tc, lane), o0);
    o1 = wmma_bf16(ap, load_b_frag(vbase + (size_t)16 * Tc + kb, Tc, lane), o1);
    o2 = wmma_bf16(ap, load_b_frag(vbase + (size_t)32 * Tc + kb, Tc, lane), o2);
    o3 = wmma_bf16(ap, load_b_frag(vbase + (size_t)48 * Tc + kb, Tc, lane), o3);
  }

  // Epilogue: normalize and write attn[b*T+t][h*64+d] bf16 for final GEMM.
  #pragma unroll
  for (int e = 0; e < 8; ++e) {
    const float inv = 1.0f / l[e];
    const int t = q0 + e + 8 * g;
    const size_t rowb = ((size_t)b * Tc + t) * Cc;
    const int cb = h * Dc + col;
    attn[rowb + cb]      = (__bf16)(o0[e] * inv);
    attn[rowb + cb + 16] = (__bf16)(o1[e] * inv);
    attn[rowb + cb + 32] = (__bf16)(o2[e] * inv);
    attn[rowb + cb + 48] = (__bf16)(o3[e] * inv);
  }
}

// ---------------------------------------------------------------------------
extern "C" void kernel_launch(void* const* d_in, const int* in_sizes, int n_in,
                              void* d_out, int out_size, void* d_ws, size_t ws_size,
                              hipStream_t stream) {
  (void)in_sizes; (void)n_in; (void)out_size; (void)ws_size;
  const float* x     = (const float*)d_in[0];
  // d_in[1] = attention_mask (all ones; unused by reference math)
  const float* w_qkv = (const float*)d_in[2];
  const float* b_qkv = (const float*)d_in[3];
  const float* w_o   = (const float*)d_in[4];
  const float* b_o   = (const float*)d_in[5];
  const float* rms_w = (const float*)d_in[6];
  float* out = (float*)d_out;

  // Workspace layout (total 96 MB):
  char* ws = (char*)d_ws;
  __bf16* xn   = (__bf16*)(ws);                          //  8 MB [4096][1024]
  __bf16* wqb  = (__bf16*)(ws + (size_t)8  * (1 << 20)); //  6 MB [3072][1024]
  __bf16* wob  = (__bf16*)(ws + (size_t)14 * (1 << 20)); //  2 MB [1024][1024]
  float*  qkvf = (float*) (ws + (size_t)16 * (1 << 20)); // 48 MB [4096][3072]
  __bf16* qhp  = (__bf16*)(ws + (size_t)64 * (1 << 20)); //  8 MB [B][H][T][D]
  __bf16* khp  = (__bf16*)(ws + (size_t)72 * (1 << 20)); //  8 MB [B][H][T][D]
  __bf16* vtp  = (__bf16*)(ws + (size_t)80 * (1 << 20)); //  8 MB [B][H][D][T]
  __bf16* attn = (__bf16*)(ws + (size_t)88 * (1 << 20)); //  8 MB [4096][1024]

  const int BT = Bc * Tc;  // 4096

  rmsnorm_bf16_kernel<<<BT, 256, 0, stream>>>(x, rms_w, xn);
  cvt_f32_bf16_kernel<<<(3 * Cc * Cc) / 1024, 256, 0, stream>>>(w_qkv, wqb, 3 * Cc * Cc);
  cvt_f32_bf16_kernel<<<(Cc * Cc) / 1024, 256, 0, stream>>>(w_o, wob, Cc * Cc);

  gemm_bf16_nt_kernel<<<dim3(BT / 128, (3 * Cc) / 64), 256, 0, stream>>>(
      xn, wqb, b_qkv, qkvf, BT, 3 * Cc, Cc);

  rope_split_kernel<<<BT, 256, 0, stream>>>(qkvf, qhp, khp, vtp);

  flash_attn_kernel<<<Bc * Hc * (Tc / 128), 256, 0, stream>>>(qhp, khp, vtp, attn);

  gemm_bf16_nt_kernel<<<dim3(BT / 128, Cc / 64), 256, 0, stream>>>(
      attn, wob, b_o, out, BT, Cc, Cc);
}
